// LSTM_MCQ_Model_64295660422003
// MI455X (gfx1250) — matrix operand order repaired and checked by reading, not compiled, not run on hardware
//
#include <hip/hip_runtime.h>
#include <hip/hip_bf16.h>
#include <stddef.h>

// ---------------------------------------------------------------------------
// Types for CDNA5 WMMA (wave32): v16bf A/B fragments, v8f accumulator.
// ---------------------------------------------------------------------------
typedef __attribute__((ext_vector_type(16))) __bf16 v16bf;
typedef __attribute__((ext_vector_type(8)))  __bf16 v8bf;
typedef __attribute__((ext_vector_type(8)))  float  v8f;

union V16 { v16bf v; v8bf h[2]; };

#define EDIM 512
#define HDIM 512
#define GDIM 2048          // 4*H
#define LROWV 65           // v8bf per LDS row (= 520 bf16, 1040 B: 16B-aligned, bank-spread)

__device__ __forceinline__ unsigned short f2bf(float x) {
  union { float f; unsigned u; } u; u.f = x;
  unsigned r = u.u + 0x7FFFu + ((u.u >> 16) & 1u);   // round-to-nearest-even
  return (unsigned short)(r >> 16);
}
__device__ __forceinline__ float sigm(float x) {
  return 1.0f / (1.0f + __expf(-x));
}
__device__ __forceinline__ float tanh_fast(float x) {
  x = fminf(15.0f, fmaxf(-15.0f, x));
  float e = __expf(2.0f * x);
  return (e - 1.0f) / (e + 1.0f);
}

// ---------------------------------------------------------------------------
// Prep: f32 -> bf16 conversions (weights + question are already in the layout
// we want; choices are regathered [b][k][t][e] -> [s=k*64+b][t][e]).
// ---------------------------------------------------------------------------
__global__ void k_cvt(const float* __restrict__ src, unsigned short* __restrict__ dst, int n) {
  int i = blockIdx.x * blockDim.x + threadIdx.x;
  int stride = gridDim.x * blockDim.x;
  for (; i < n; i += stride) dst[i] = f2bf(src[i]);
}

__global__ void k_cvt_choices(const float* __restrict__ src, unsigned short* __restrict__ dst) {
  const int n = 256 * 64 * 512;
  int i = blockIdx.x * blockDim.x + threadIdx.x;
  int stride = gridDim.x * blockDim.x;
  for (; i < n; i += stride) {
    int e = i & 511;
    int r = i >> 9;
    int t = r & 63;
    int s = r >> 6;            // 0..255
    int k = s >> 6, b = s & 63;
    dst[i] = f2bf(src[(((size_t)(b * 4 + k)) * 64 + t) * 512 + e]);
  }
}

__global__ void k_bias(const float* __restrict__ bi, const float* __restrict__ bh,
                       float* __restrict__ bs) {
  int i = blockIdx.x * blockDim.x + threadIdx.x;
  if (i < GDIM) bs[i] = bi[i] + bh[i];
}

// ---------------------------------------------------------------------------
// Fused LSTM recurrence. One workgroup = 16 batch rows, 16 waves x 32 lanes.
// Wave w owns hidden columns [w*32, w*32+32): 8 accumulator tiles, one pair
// per gate region so i/f/g/o align elementwise in registers.
// gates_t = x_t*Wih^T + h_{t-1}*Whh^T + b   (both K=512 loops on WMMA bf16)
// ---------------------------------------------------------------------------
__global__ __launch_bounds__(512, 1)
void k_lstm(const unsigned short* __restrict__ Xq, const unsigned short* __restrict__ Xc,
            const unsigned short* __restrict__ Wih_g, const unsigned short* __restrict__ Whh_g,
            const float* __restrict__ bias, float* __restrict__ Hq, float* __restrict__ Hc) {
  __shared__ v8bf hbufv[16 * LROWV];                 // h_{t-1} as bf16, padded rows
  unsigned short* hb = (unsigned short*)hbufv;       // scalar store view (same LDS)

  const int wg   = blockIdx.x;
  const bool isQ = (wg < 4);
  const int T    = isQ ? 256 : 64;
  const int seq0 = isQ ? wg * 16 : (wg - 4) * 16;
  const unsigned short* X = isQ ? Xq : Xc;
  float* Hout             = isQ ? Hq : Hc;

  const int tid   = threadIdx.x;
  const int lane  = tid & 31;
  const int wave  = tid >> 5;      // 0..15
  const int lhalf = lane >> 4;     // 0/1
  const int lmod  = lane & 15;

  {
    union { unsigned short u[8]; v8bf v; } z;
#pragma unroll
    for (int i = 0; i < 8; ++i) z.u[i] = 0;
    for (int i = tid; i < 16 * LROWV; i += 512) hbufv[i] = z.v;   // h0 = 0
  }
  __syncthreads();

  // A-matrix row for this lane (rows 0..15 of the batch tile)
  const unsigned short* xrow = X + (size_t)(seq0 + lmod) * T * EDIM;

  // Per-tile B offsets (elements): row (r*512 + wave*32 + j*16 + lmod),
  // plus the lane-half K offset. k0 is added per iteration.
  size_t wofs[4][2];
  float  bval[4][2];
#pragma unroll
  for (int r = 0; r < 4; ++r)
#pragma unroll
    for (int j = 0; j < 2; ++j) {
      const int grow = r * 512 + wave * 32 + j * 16 + lmod;
      wofs[r][j] = (size_t)grow * EDIM + lhalf * 8;
      bval[r][j] = bias[grow];
    }

  float cst[2][8];
#pragma unroll
  for (int j = 0; j < 2; ++j)
#pragma unroll
    for (int v = 0; v < 8; ++v) cst[j][v] = 0.0f;

  float hval[2][8];
#pragma unroll
  for (int j = 0; j < 2; ++j)
#pragma unroll
    for (int v = 0; v < 8; ++v) hval[j][v] = 0.0f;

  // Opaque zero offset, redefined every timestep: keeps weight-fragment loads
  // inside the t-loop (no LICM -> no spill) while PRESERVING pointer
  // provenance, so they stay global_load_b128 (not flat).
  unsigned int tofs = 0;

  for (int t = 0; t < T; ++t) {
    asm volatile("" : "+s"(tofs));
    const unsigned short* Wih = Wih_g + tofs;
    const unsigned short* Whh = Whh_g + tofs;

    v8f acc[4][2];
#pragma unroll
    for (int r = 0; r < 4; ++r)
#pragma unroll
      for (int j = 0; j < 2; ++j)
#pragma unroll
        for (int v = 0; v < 8; ++v) acc[r][j][v] = 0.0f;   // bias added later

    // ---- x_t * Wih^T : K = 512 over embedding dim, A from global bf16 ----
    const unsigned short* xt = xrow + (size_t)t * EDIM;
    for (int k0 = 0; k0 < EDIM; k0 += 32) {
      V16 a;
      const unsigned short* ap = xt + k0 + lhalf * 8;
      a.h[0] = *(const v8bf*)(const void*)(ap);
      a.h[1] = *(const v8bf*)(const void*)(ap + 16);
      V16 bm[8];
#pragma unroll
      for (int r = 0; r < 4; ++r)
#pragma unroll
        for (int j = 0; j < 2; ++j) {
          const unsigned short* wp = Wih + wofs[r][j] + k0;
          bm[r * 2 + j].h[0] = *(const v8bf*)(const void*)(wp);
          bm[r * 2 + j].h[1] = *(const v8bf*)(const void*)(wp + 16);
        }
#pragma unroll
      for (int r = 0; r < 4; ++r)
#pragma unroll
        for (int j = 0; j < 2; ++j)
          acc[r][j] = __builtin_amdgcn_wmma_f32_16x16x32_bf16(
              false, a.v, false, bm[r * 2 + j].v, (short)0, acc[r][j], false, false);
    }

    // ---- h_{t-1} * Whh^T : K = 512 over hidden dim, A from LDS ----
    for (int k0 = 0; k0 < HDIM; k0 += 32) {
      V16 a;
      const int hv = lmod * LROWV + (k0 >> 3) + lhalf;
      a.h[0] = hbufv[hv];        // ds_load_b128
      a.h[1] = hbufv[hv + 2];    // +16 bf16 elements
      V16 bm[8];
#pragma unroll
      for (int r = 0; r < 4; ++r)
#pragma unroll
        for (int j = 0; j < 2; ++j) {
          const unsigned short* wp = Whh + wofs[r][j] + k0;
          bm[r * 2 + j].h[0] = *(const v8bf*)(const void*)(wp);
          bm[r * 2 + j].h[1] = *(const v8bf*)(const void*)(wp + 16);
        }
#pragma unroll
      for (int r = 0; r < 4; ++r)
#pragma unroll
        for (int j = 0; j < 2; ++j)
          acc[r][j] = __builtin_amdgcn_wmma_f32_16x16x32_bf16(
              false, a.v, false, bm[r * 2 + j].v, (short)0, acc[r][j], false, false);
    }

    // ---- gate nonlinearity + c/h update, fully in registers ----
#pragma unroll
    for (int j = 0; j < 2; ++j)
#pragma unroll
      for (int v = 0; v < 8; ++v) {
        float iv = sigm(acc[0][j][v] + bval[0][j]);
        float fv = sigm(acc[1][j][v] + bval[1][j]);
        float gv = tanh_fast(acc[2][j][v] + bval[2][j]);
        float ov = sigm(acc[3][j][v] + bval[3][j]);
        float c  = fv * cst[j][v] + iv * gv;
        cst[j][v]  = c;
        hval[j][v] = ov * tanh_fast(c);
      }

    __syncthreads();   // all waves finished reading h_{t-1} from LDS
#pragma unroll
    for (int j = 0; j < 2; ++j) {
      const int col = wave * 32 + j * 16 + lmod;
#pragma unroll
      for (int v = 0; v < 8; ++v) {
        const int row = v + lhalf * 8;    // C-layout: VGPR v, lane half
        hb[row * (LROWV * 8) + col] = f2bf(hval[j][v]);
      }
    }
    __syncthreads();   // h_t visible for next step
  }

  // final hidden state, f32
#pragma unroll
  for (int j = 0; j < 2; ++j) {
    const int col = wave * 32 + j * 16 + lmod;
#pragma unroll
    for (int v = 0; v < 8; ++v) {
      const int row = v + lhalf * 8;
      Hout[(size_t)(seq0 + row) * HDIM + col] = hval[j][v];
    }
  }
}

// ---------------------------------------------------------------------------
// logits[b,k] = <Hc[k*64+b], Hq[b]>, then softmax over b (axis 0) per k.
// ---------------------------------------------------------------------------
__global__ __launch_bounds__(256)
void k_logits(const float* __restrict__ Hq, const float* __restrict__ Hc,
              float* __restrict__ out) {
  __shared__ float lg[256];
  const int tid = threadIdx.x;
  const int b = tid >> 2, k = tid & 3;
  const float* q = Hq + (size_t)b * HDIM;
  const float* c = Hc + (size_t)(k * 64 + b) * HDIM;
  float s = 0.0f;
  for (int e = 0; e < HDIM; ++e) s += q[e] * c[e];
  lg[b * 4 + k] = s;
  __syncthreads();
  float mx = -1e30f;
  for (int bb = 0; bb < 64; ++bb) mx = fmaxf(mx, lg[bb * 4 + k]);
  float sum = 0.0f;
  for (int bb = 0; bb < 64; ++bb) sum += __expf(lg[bb * 4 + k] - mx);
  out[b * 4 + k] = __expf(s - mx) / sum;
}

// ---------------------------------------------------------------------------
extern "C" void kernel_launch(void* const* d_in, const int* in_sizes, int n_in,
                              void* d_out, int out_size, void* d_ws, size_t ws_size,
                              hipStream_t stream) {
  const float* q     = (const float*)d_in[0];   // [64,256,512]
  const float* ch    = (const float*)d_in[1];   // [64,4,64,512]
  const float* Wih_f = (const float*)d_in[2];   // [2048,512]
  const float* Whh_f = (const float*)d_in[3];   // [2048,512]
  const float* bih   = (const float*)d_in[4];   // [2048]
  const float* bhh   = (const float*)d_in[5];   // [2048]

  char* ws = (char*)d_ws;
  size_t off = 0;
  auto carve = [&](size_t bytes) -> char* {
    char* p = ws + off;
    off += (bytes + 255) & ~(size_t)255;
    return p;
  };
  unsigned short* Xq   = (unsigned short*)carve((size_t)64 * 256 * 512 * 2);  // 16 MB
  unsigned short* Xc   = (unsigned short*)carve((size_t)256 * 64 * 512 * 2);  // 16 MB
  unsigned short* Wih  = (unsigned short*)carve((size_t)GDIM * EDIM * 2);     //  2 MB
  unsigned short* Whh  = (unsigned short*)carve((size_t)GDIM * HDIM * 2);     //  2 MB
  float*          bsum = (float*)carve((size_t)GDIM * 4);
  float*          Hq   = (float*)carve((size_t)64 * HDIM * 4);
  float*          Hc   = (float*)carve((size_t)256 * HDIM * 4);
  (void)ws_size; (void)in_sizes; (void)n_in; (void)out_size;

  // prep: bf16 conversions + bias combine
  k_cvt<<<4096, 256, 0, stream>>>(q, Xq, 64 * 256 * 512);
  k_cvt_choices<<<4096, 256, 0, stream>>>(ch, Xc);
  k_cvt<<<2048, 256, 0, stream>>>(Wih_f, Wih, GDIM * EDIM);
  k_cvt<<<2048, 256, 0, stream>>>(Whh_f, Whh, GDIM * HDIM);
  k_bias<<<8, 256, 0, stream>>>(bih, bhh, bsum);

  // fused recurrence: 4 question WGs + 16 choice WGs
  k_lstm<<<20, 512, 0, stream>>>(Xq, Xc, Wih, Whh, bsum, Hq, Hc);

  // logits + axis-0 softmax
  k_logits<<<1, 256, 0, stream>>>(Hq, Hc, (float*)d_out);
}
